// MultiHeadAttention_37512244363503
// MI455X (gfx1250) — matrix operand order repaired and checked by reading, not compiled, not run on hardware
//
#include <hip/hip_runtime.h>
#include <stdint.h>
#include <stddef.h>

// ---------------------------------------------------------------------------
// Types for CDNA5 WMMA (gfx1250, wave32)
// ---------------------------------------------------------------------------
typedef __bf16 bf16;
typedef bf16  v8bf  __attribute__((ext_vector_type(8)));
typedef bf16  v16bf __attribute__((ext_vector_type(16)));
typedef float v8f   __attribute__((ext_vector_type(8)));

// Problem constants (match reference)
#define BB   4
#define SEQ  1024     // N == L
#define DIM  1024     // D
#define NH   16       // heads
#define DHD  64       // per-head channels

__device__ __forceinline__ v8f vzero8() {
    v8f z;
#pragma unroll
    for (int i = 0; i < 8; ++i) z[i] = 0.0f;
    return z;
}

// A-fragment (16x32 bf16, MxK): lane = row (lane&15); this lane's 16 values are
// K = kb..kb+15 contiguous, kb = (lane>>4)*16.  p must point at rowBase + k0 + kb.
__device__ __forceinline__ v16bf load_afrag(const bf16* p) {
    v8bf lo = *reinterpret_cast<const v8bf*>(p);
    v8bf hi = *reinterpret_cast<const v8bf*>(p + 8);
    return __builtin_shufflevector(lo, hi, 0, 1, 2, 3, 4, 5, 6, 7,
                                           8, 9, 10, 11, 12, 13, 14, 15);
}

// B-fragment (32x16 bf16, KxN): lane = column (lane&15); this lane's 16 values
// are K = kb..kb+15 contiguous, kb = (lane>>4)*16.  B stored K-contiguous per
// column ("B^T row-major").
__device__ __forceinline__ v16bf load_bfrag(const bf16* p) {
    return *reinterpret_cast<const v16bf*>(p);
}

__device__ __forceinline__ v8f wmma_bf16(v16bf a, v16bf b, v8f c) {
    // v_wmma_f32_16x16x32_bf16
    return __builtin_amdgcn_wmma_f32_16x16x32_bf16(
        /*neg_a=*/false, a, /*neg_b=*/false, b,
        /*c_mod=*/(short)0, c, /*reuse_a=*/false, /*reuse_b=*/false);
}

__device__ __forceinline__ float redmax16(float v) {
    v = fmaxf(v, __shfl_xor(v, 1, 32));
    v = fmaxf(v, __shfl_xor(v, 2, 32));
    v = fmaxf(v, __shfl_xor(v, 4, 32));
    v = fmaxf(v, __shfl_xor(v, 8, 32));
    return v;
}
__device__ __forceinline__ float redsum16(float v) {
    v += __shfl_xor(v, 1, 32);
    v += __shfl_xor(v, 2, 32);
    v += __shfl_xor(v, 4, 32);
    v += __shfl_xor(v, 8, 32);
    return v;
}

// ---------------------------------------------------------------------------
// Elementwise f32 -> bf16
// ---------------------------------------------------------------------------
__global__ __launch_bounds__(256) void f2bf_kernel(const float* __restrict__ x,
                                                   bf16* __restrict__ y, int n) {
    int i = blockIdx.x * 256 + threadIdx.x;
    if (i < n) y[i] = (bf16)x[i];
}

// Wt[n][k] = (bf16) W[k][n]   (DIM x DIM)
__global__ __launch_bounds__(256) void wtrans_kernel(const float* __restrict__ W,
                                                     bf16* __restrict__ Wt) {
    int i = blockIdx.x * 256 + threadIdx.x;   // 0 .. DIM*DIM-1
    int n = i >> 10;
    int k = i & (DIM - 1);
    Wt[i] = (bf16)W[(size_t)k * DIM + n];
}

// ---------------------------------------------------------------------------
// WMMA GEMM:  C[M=4096][DIM] = A(bf16, row-major MxK=DIM) @ Wt^T
//   Wt stored as Wt[n][k] (column of W contiguous over k).
// Block = 8 waves (4 M-rows x 2 N-cols); per-wave tile 32(M) x 64(N).
// Block tile 128 x 128.  B (weights) staged in LDS via CDNA5 async-to-LDS
// loads, double-buffered per 32-wide K step; A fragments come straight from
// global (L2-resident).  LDS rows padded to 40 bf16 -> conflict-free
// ds_load_b128 B-fragment reads.
//   mode 0: write bf16 to headed layout (B,H,S,DH)
//   mode 2: write bf16 to transposed-head layout (B,H,DH,S)
//   mode 3: write f32 row-major + bias (final projection)
// ---------------------------------------------------------------------------
__global__ __launch_bounds__(256) void gemm_kernel(const bf16* __restrict__ A,
                                                   const bf16* __restrict__ Wt,
                                                   const float* __restrict__ bias,
                                                   void* __restrict__ outp,
                                                   int mode) {
    __shared__ bf16 bbuf[2][128][40];   // [buf][n in block][k], row padded to 40

    const int tid  = threadIdx.x;
    const int lane = tid & 31;
    const int wave = tid >> 5;
    const int cl = lane & 15;
    const int hi = lane >> 4;
    const int koff = hi * 8;    // unused (A-fragment uses kb16)
    const int kb16 = hi * 16;   // B-fragment K offset

    const int m0   = blockIdx.x * 128 + (wave & 3) * 32;  // wave M base
    const int nblk = blockIdx.y * 128;                    // block N base
    const int nloc = (wave >> 2) * 64;                    // wave N base in block

    const bf16* arow0 = A + (size_t)(m0 + cl) * DIM;
    const bf16* arow1 = A + (size_t)(m0 + 16 + cl) * DIM;

    const uint64_t wbase = (uint64_t)(uintptr_t)Wt;

    // Cooperative async staging of the 128x32 B tile (8 KB): 512 x 16-byte
    // chunks, 2 per thread.  chunk c: row = c>>2, sub = c&3 (8 bf16 each).
    const int r0 = tid >> 2,         s0c = tid & 3;
    const int r1 = (tid + 256) >> 2, s1c = (tid + 256) & 3;

    auto issue_b = [&](int k, int nb) {
        uint32_t vo0 = (uint32_t)((((nblk + r0) * DIM) + k + s0c * 8) * 2);
        uint32_t ld0 = (uint32_t)(uintptr_t)&bbuf[nb][r0][s0c * 8];
        asm volatile("global_load_async_to_lds_b128 %0, %1, %2"
                     :: "v"(ld0), "v"(vo0), "s"(wbase) : "memory");
        uint32_t vo1 = (uint32_t)((((nblk + r1) * DIM) + k + s1c * 8) * 2);
        uint32_t ld1 = (uint32_t)(uintptr_t)&bbuf[nb][r1][s1c * 8];
        asm volatile("global_load_async_to_lds_b128 %0, %1, %2"
                     :: "v"(ld1), "v"(vo1), "s"(wbase) : "memory");
    };

    v8f acc[2][4];
#pragma unroll
    for (int ms = 0; ms < 2; ++ms)
#pragma unroll
        for (int t = 0; t < 4; ++t) acc[ms][t] = vzero8();

    // prologue: fill buffer 0
    issue_b(0, 0);
    asm volatile("s_wait_asynccnt 0" ::: "memory");
    __syncthreads();

    for (int k = 0; k < DIM; k += 32) {
        const int cur = (k >> 5) & 1;
        if (k + 32 < DIM) issue_b(k + 32, cur ^ 1);   // prefetch next tile

        if (k + 64 < DIM) __builtin_prefetch(arow0 + k + 64, 0, 3);
        v16bf a0 = load_afrag(arow0 + k + kb16);
        v16bf a1 = load_afrag(arow1 + k + kb16);
#pragma unroll
        for (int t = 0; t < 4; ++t) {
            v16bf b = load_bfrag(&bbuf[cur][nloc + t * 16 + cl][kb16]);
            acc[0][t] = wmma_bf16(a0, b, acc[0][t]);
            acc[1][t] = wmma_bf16(a1, b, acc[1][t]);
        }

        asm volatile("s_wait_asynccnt 0" ::: "memory");  // next tile landed
        __syncthreads();                                  // visible block-wide
    }

    // C layout: VGPR r, lane -> row = r + 8*hi, col = cl (within 16x16 tile)
    if (mode == 0) {
        bf16* o = (bf16*)outp;
#pragma unroll
        for (int ms = 0; ms < 2; ++ms)
#pragma unroll
            for (int t = 0; t < 4; ++t)
#pragma unroll
                for (int r = 0; r < 8; ++r) {
                    int gm = m0 + ms * 16 + r + 8 * hi;
                    int gc = nloc + nblk + t * 16 + cl;
                    int b = gm >> 10, s = gm & (SEQ - 1);
                    int h = gc >> 6, dh = gc & (DHD - 1);
                    o[(((size_t)(b * NH + h)) * SEQ + s) * DHD + dh] = (bf16)acc[ms][t][r];
                }
    } else if (mode == 2) {
        bf16* o = (bf16*)outp;
#pragma unroll
        for (int ms = 0; ms < 2; ++ms)
#pragma unroll
            for (int t = 0; t < 4; ++t)
#pragma unroll
                for (int r = 0; r < 8; ++r) {
                    int gm = m0 + ms * 16 + r + 8 * hi;
                    int gc = nloc + nblk + t * 16 + cl;
                    int b = gm >> 10, s = gm & (SEQ - 1);
                    int h = gc >> 6, dh = gc & (DHD - 1);
                    o[(((size_t)(b * NH + h)) * DHD + dh) * SEQ + s] = (bf16)acc[ms][t][r];
                }
    } else { // mode 3
        float* o = (float*)outp;
#pragma unroll
        for (int ms = 0; ms < 2; ++ms)
#pragma unroll
            for (int t = 0; t < 4; ++t)
#pragma unroll
                for (int r = 0; r < 8; ++r) {
                    int gm = m0 + ms * 16 + r + 8 * hi;
                    int gc = nloc + nblk + t * 16 + cl;
                    o[(size_t)gm * DIM + gc] = acc[ms][t][r] + bias[gc];
                }
    }
}

// ---------------------------------------------------------------------------
// Flash attention. One wave handles one 16-row Q tile of one (b,h).
// Loops over L in chunks of 32: S = Q K^T (4 wmma), online softmax,
// P reshaped C->A layout through wave-private LDS, O += P V (4 wmma).
// Output written bf16 to (B, N, D) row-major (D index = h*64 + dh).
// ---------------------------------------------------------------------------
__global__ __launch_bounds__(128) void attn_kernel(const bf16* __restrict__ qb,
                                                   const bf16* __restrict__ kb,
                                                   const bf16* __restrict__ vtb,
                                                   const uint8_t* __restrict__ mask,
                                                   bf16* __restrict__ ob) {
    __shared__ bf16 p_lds[4][16][32];   // per-wave 16x32 P tile

    const int lane = threadIdx.x & 31;
    const int wave = threadIdx.x >> 5;
    const int cl = lane & 15;
    const int hi = lane >> 4;
    const int koff = hi * 8;
    const int kb16 = hi * 16;

    const int b = blockIdx.z;
    const int h = blockIdx.y;
    const int q0 = (blockIdx.x * 4 + wave) * 16;

    const bf16* qh = qb + (((size_t)(b * NH + h)) * SEQ + q0) * DHD;
    const bf16* kh = kb + ((size_t)(b * NH + h)) * SEQ * DHD;
    const bf16* vh = vtb + ((size_t)(b * NH + h)) * DHD * SEQ;
    const uint8_t* mk = mask + (size_t)b * SEQ;

    // Q fragments (row = cl), two K-steps of 32 covering DH=64
    v16bf qa[2];
#pragma unroll
    for (int s = 0; s < 2; ++s)
        qa[s] = load_afrag(qh + cl * DHD + s * 32 + kb16);

    const float scale = 0.125f;            // DH^-0.5
    const float NEG = -3.4028234e38f;

    float m_run[8], l_run[8];
    v8f oacc[4];
#pragma unroll
    for (int r = 0; r < 8; ++r) { m_run[r] = NEG; l_run[r] = 0.0f; }
#pragma unroll
    for (int t = 0; t < 4; ++t) oacc[t] = vzero8();

    for (int l0 = 0; l0 < SEQ; l0 += 32) {
        // ---- S = Q K^T for two 16-col tiles ----
        v8f s0 = vzero8(), s1 = vzero8();
#pragma unroll
        for (int s = 0; s < 2; ++s) {
            const bf16* kp0 = kh + (size_t)(l0 + cl) * DHD + s * 32 + kb16;
            const bf16* kp1 = kh + (size_t)(l0 + 16 + cl) * DHD + s * 32 + kb16;
            s0 = wmma_bf16(qa[s], load_bfrag(kp0), s0);
            s1 = wmma_bf16(qa[s], load_bfrag(kp1), s1);
        }

        // ---- scale + pad mask (lane owns one L column per tile) ----
        const bool mA = mk[l0 + cl] != 0;
        const bool mB = mk[l0 + 16 + cl] != 0;
#pragma unroll
        for (int r = 0; r < 8; ++r) {
            s0[r] = mA ? NEG : s0[r] * scale;
            s1[r] = mB ? NEG : s1[r] * scale;
        }

        // ---- online softmax (row reductions across the 16-lane half) ----
        float cor[8];
#pragma unroll
        for (int r = 0; r < 8; ++r) {
            float tmax = redmax16(fmaxf(s0[r], s1[r]));
            float mn = fmaxf(m_run[r], tmax);
            cor[r] = __expf(m_run[r] - mn);
            m_run[r] = mn;
            float p0 = __expf(s0[r] - mn);
            float p1 = __expf(s1[r] - mn);
            l_run[r] = l_run[r] * cor[r] + redsum16(p0 + p1);
            // stash P into LDS: row = r + 8*hi, cols cl and 16+cl
            p_lds[wave][r + 8 * hi][cl]      = (bf16)p0;
            p_lds[wave][r + 8 * hi][16 + cl] = (bf16)p1;
        }
#pragma unroll
        for (int t = 0; t < 4; ++t)
#pragma unroll
            for (int r = 0; r < 8; ++r) oacc[t][r] *= cor[r];

        // make cross-lane LDS writes visible before A-fragment reload
        // (DS ops are in-order within a wave; wait drains data return)
        asm volatile("s_wait_dscnt 0" ::: "memory");

        // ---- O += P V  (P: 16x32 A-frag, V^T rows contiguous over L) ----
        v16bf pa = load_afrag(&p_lds[wave][cl][kb16]);
#pragma unroll
        for (int t = 0; t < 4; ++t) {
            const bf16* vp = vh + (size_t)(t * 16 + cl) * SEQ + l0 + kb16;
            oacc[t] = wmma_bf16(pa, load_bfrag(vp), oacc[t]);
        }
    }

    // ---- finalize: divide by row sums, store bf16 to (B, N, D) ----
    float inv[8];
#pragma unroll
    for (int r = 0; r < 8; ++r) inv[r] = 1.0f / l_run[r];
#pragma unroll
    for (int t = 0; t < 4; ++t)
#pragma unroll
        for (int r = 0; r < 8; ++r) {
            int row = q0 + r + 8 * hi;
            int col = h * DHD + t * 16 + cl;
            ob[((size_t)(b * SEQ + row)) * DIM + col] = (bf16)(oacc[t][r] * inv[r]);
        }
}

// ---------------------------------------------------------------------------
// Host launcher
// ---------------------------------------------------------------------------
extern "C" void kernel_launch(void* const* d_in, const int* in_sizes, int n_in,
                              void* d_out, int out_size, void* d_ws, size_t ws_size,
                              hipStream_t stream) {
    (void)in_sizes; (void)n_in; (void)out_size; (void)ws_size;

    const float*   x_q  = (const float*)d_in[0];
    const float*   x_kv = (const float*)d_in[1];
    const uint8_t* pad  = (const uint8_t*)d_in[2];
    const float*   Wq   = (const float*)d_in[3];
    const float*   Wk   = (const float*)d_in[4];
    const float*   Wv   = (const float*)d_in[5];
    const float*   Wo   = (const float*)d_in[6];
    const float*   bo   = (const float*)d_in[7];

    const size_t NX = (size_t)BB * SEQ * DIM;   // 4,194,304 elements
    const size_t NW = (size_t)DIM * DIM;        // 1,048,576 elements

    char* p = (char*)d_ws;
    bf16* xq_b  = (bf16*)p; p += NX * sizeof(bf16);
    bf16* xkv_b = (bf16*)p; p += NX * sizeof(bf16);
    bf16* wq_t  = (bf16*)p; p += NW * sizeof(bf16);
    bf16* wk_t  = (bf16*)p; p += NW * sizeof(bf16);
    bf16* wv_t  = (bf16*)p; p += NW * sizeof(bf16);
    bf16* wo_t  = (bf16*)p; p += NW * sizeof(bf16);
    bf16* qh    = (bf16*)p; p += NX * sizeof(bf16);
    bf16* kh    = (bf16*)p; p += NX * sizeof(bf16);
    bf16* vt    = (bf16*)p; p += NX * sizeof(bf16);
    bf16* obuf  = (bf16*)p; p += NX * sizeof(bf16);

    // 1) precision conversion
    f2bf_kernel<<<(int)(NX / 256), 256, 0, stream>>>(x_q, xq_b, (int)NX);
    f2bf_kernel<<<(int)(NX / 256), 256, 0, stream>>>(x_kv, xkv_b, (int)NX);
    wtrans_kernel<<<(int)(NW / 256), 256, 0, stream>>>(Wq, wq_t);
    wtrans_kernel<<<(int)(NW / 256), 256, 0, stream>>>(Wk, wk_t);
    wtrans_kernel<<<(int)(NW / 256), 256, 0, stream>>>(Wv, wv_t);
    wtrans_kernel<<<(int)(NW / 256), 256, 0, stream>>>(Wo, wo_t);

    // 2) projections (M = B*SEQ = 4096), block tile 128x128
    dim3 ggrid(4096 / 128, DIM / 128);
    gemm_kernel<<<ggrid, 256, 0, stream>>>(xq_b,  wq_t, nullptr, qh,   0);
    gemm_kernel<<<ggrid, 256, 0, stream>>>(xkv_b, wk_t, nullptr, kh,   0);
    gemm_kernel<<<ggrid, 256, 0, stream>>>(xkv_b, wv_t, nullptr, vt,   2);

    // 3) flash attention: grid (N/64, H, B), 4 waves per block
    attn_kernel<<<dim3(SEQ / 64, NH, BB), 128, 0, stream>>>(qh, kh, vt, pad, obuf);

    // 4) output projection + bias (f32 out)
    gemm_kernel<<<ggrid, 256, 0, stream>>>(obuf, wo_t, bo, d_out, 3);
}
